// SBAttention_8976481649328
// MI455X (gfx1250) — matrix operand order repaired and checked by reading, not compiled, not run on hardware
//
#include <hip/hip_runtime.h>
#include <hip/hip_bf16.h>
#include <math.h>
#include <stdint.h>

#define DEV static __device__ __forceinline__
#define USE_ASYNC 1

typedef __attribute__((ext_vector_type(16))) __bf16 v16bf;
typedef __attribute__((ext_vector_type(8)))  __bf16 v8bf;
typedef __attribute__((ext_vector_type(8)))  float  v8f;

// ---- problem constants (fixed by reference) ----
constexpr int BATCH = 4, SEQ = 4096, HEADS = 8, EDIM = 64, MFEAT = 128;
constexpr int NHASH = 2, BSZ = 128, BHD = BATCH * HEADS, NBUK = SEQ / BSZ;
constexpr int DMODEL = HEADS * EDIM;                 // 512
constexpr float TEMP = 0.125f;                       // 1/sqrt(64)
constexpr float SQRT_TEMP = 0.35355339059327373f;
constexpr float HALF_LOG_M = 2.4260151319598084f;    // 0.5*ln(128)

// ---- helpers ----
DEV __bf16 f2bf(float f) {
  unsigned u = __builtin_bit_cast(unsigned, f);
  unsigned r = u + 0x7FFFu + ((u >> 16) & 1u);       // RNE truncate to bf16
  unsigned short h = (unsigned short)(r >> 16);
  return __builtin_bit_cast(__bf16, h);
}

DEV v8f vzero() {
  v8f z;
#pragma unroll
  for (int e = 0; e < 8; ++e) z[e] = 0.f;
  return z;
}

DEV v8f wmma_bf16(v16bf a, v16bf b, v8f c) {
  return __builtin_amdgcn_wmma_f32_16x16x32_bf16(false, a, false, b, (short)0, c,
                                                 false, false);
}

// A fragment from row-major bf16 (rows x K), base pre-offset to (row0, k0).
// Lane layout k = 16h + s (one contiguous 16-elem run, same as B) -> 2x b128 loads.
DEV v16bf frag_a_row(const __bf16* base, int stride, int lane) {
  int r = lane & 15, h = (lane >> 4) & 1;
  const __bf16* p = base + r * stride + 16 * h;
  v8bf lo = *(const v8bf*)p;
  v8bf hi = *(const v8bf*)(p + 8);
  return __builtin_shufflevector(lo, hi, 0, 1, 2, 3, 4, 5, 6, 7, 8, 9, 10, 11,
                                 12, 13, 14, 15);
}

// B fragment where logical B(K x N) is the transpose of a row-major (N x K)
// array: B[k][c] = src[c*stride + k]. One contiguous 16-elem run per lane.
DEV v16bf frag_bt_row(const __bf16* base, int stride, int lane) {
  int c = lane & 15, h = (lane >> 4) & 1;
  return *(const v16bf*)(base + c * stride + 16 * h);
}

// generic (gather) fragment loaders; functor returns __bf16
template <typename F> DEV v16bf frag_a(F f, int lane) {
  int r = lane & 15, h = (lane >> 4) & 1;
  v16bf o;
#pragma unroll
  for (int i = 0; i < 4; ++i) {
    int k0 = 2 * i + 16 * h;
    o[2 * i] = f(r, k0);
    o[2 * i + 1] = f(r, k0 + 1);
    o[8 + 2 * i] = f(r, k0 + 8);
    o[8 + 2 * i + 1] = f(r, k0 + 9);
  }
  return o;
}
template <typename F> DEV v16bf frag_b(F f, int lane) {
  int c = lane & 15, h = (lane >> 4) & 1;
  v16bf o;
#pragma unroll
  for (int i = 0; i < 8; ++i) {
    o[2 * i] = f(16 * h + 2 * i, c);
    o[2 * i + 1] = f(16 * h + 2 * i + 1, c);
  }
  return o;
}
template <typename F> DEV void store_cd(const v8f& d, F f, int lane) {
  int n = lane & 15, h = (lane >> 4) & 1;
#pragma unroll
  for (int v = 0; v < 8; ++v) f(v + 8 * h, n, d[v]);
}

// order-preserving float<->uint key for atomic max over signed floats
DEV unsigned mkey(float f) {
  unsigned b = __builtin_bit_cast(unsigned, f);
  return (b & 0x80000000u) ? ~b : (b | 0x80000000u);
}
DEV float unkey(unsigned k) {
  unsigned b = (k & 0x80000000u) ? (k & 0x7FFFFFFFu) : ~k;
  return __builtin_bit_cast(float, b);
}

#if USE_ASYNC
DEV void async16(unsigned ldsOff, unsigned long long gaddr) {
  asm volatile("global_load_async_to_lds_b128 %0, %1, off"
               :: "v"(ldsOff), "v"(gaddr)
               : "memory");
}
DEV void wait_async() {
#if __has_builtin(__builtin_amdgcn_s_wait_asynccnt)
  __builtin_amdgcn_s_wait_asynccnt(0);
#else
  asm volatile("s_wait_asynccnt 0x0" ::: "memory");
#endif
}
#endif

// ================= kernels =================

__global__ void k_init(unsigned* maxq2, unsigned* maxk2, unsigned* kstab) {
  int t = threadIdx.x;
  if (t < BHD) { maxq2[t] = 0u; maxk2[t] = 0u; kstab[t] = 0u; }
}

// convert weights to bf16 once
__global__ void k_wconv(const float* __restrict__ projW,
                        const float* __restrict__ outW, __bf16* wbf,
                        __bf16* wobf) {
  int tid = blockIdx.x * blockDim.x + threadIdx.x;
  if (tid < MFEAT * EDIM) wbf[tid] = f2bf(projW[tid]);
  if (tid < EDIM * DMODEL) wobf[tid] = f2bf(outW[tid]);
}

// split heads; f32 copies (for hashing) + bf16 copies (for WMMA); norms
__global__ void k_split(const float* __restrict__ q, const float* __restrict__ k,
                        const float* __restrict__ v, float* q_s, float* k_s,
                        __bf16* qbf, __bf16* kbf, __bf16* vbf, float* qn2,
                        float* kn2, unsigned* maxq2, unsigned* maxk2) {
  int tid = blockIdx.x * blockDim.x + threadIdx.x;
  if (tid >= BHD * SEQ) return;
  int bh = tid / SEQ, n = tid % SEQ;
  int b = bh >> 3, hh = bh & 7;
  size_t src = ((size_t)b * SEQ + n) * DMODEL + hh * EDIM;
  size_t dst = ((size_t)bh * SEQ + n) * EDIM;
  float sq = 0.f, sk = 0.f;
#pragma unroll 4
  for (int e = 0; e < EDIM; ++e) {
    float qv = q[src + e], kv = k[src + e], vv = v[src + e];
    q_s[dst + e] = qv; k_s[dst + e] = kv;
    qbf[dst + e] = f2bf(qv); kbf[dst + e] = f2bf(kv); vbf[dst + e] = f2bf(vv);
    sq += qv * qv; sk += kv * kv;
  }
  qn2[tid] = sq; kn2[tid] = sk;
  atomicMax(&maxq2[bh], __float_as_uint(sq));
  atomicMax(&maxk2[bh], __float_as_uint(sk));
}

// E2LSH hashes with xbox+ extension
__global__ void k_hashes(const float* q_s, const float* k_s, const float* qn2,
                         const float* kn2, const unsigned* maxq2,
                         const unsigned* maxk2, const float* __restrict__ alpha,
                         const float* __restrict__ beta, float* q_hash,
                         float* k_hash) {
  int tid = blockIdx.x * blockDim.x + threadIdx.x;
  if (tid >= BHD * SEQ) return;
  int bh = tid / SEQ, n = tid % SEQ;
  float M2 = __uint_as_float(maxq2[bh]) + __uint_as_float(maxk2[bh]);
  float extq = sqrtf(fmaxf(M2 - qn2[tid], 0.f));
  float extk = sqrtf(fmaxf(M2 - kn2[tid], 0.f));
  size_t row = ((size_t)bh * SEQ + n) * EDIM;
#pragma unroll
  for (int nh = 0; nh < NHASH; ++nh) {
    float hq = beta[nh], hk = beta[nh];
    for (int e = 0; e < EDIM; ++e) {
      float a = alpha[e * NHASH + nh];
      hq += q_s[row + e] * a;
      hk += k_s[row + e] * a;
    }
    hq += extq * alpha[EDIM * NHASH + nh];
    hk += extk * alpha[(EDIM + 1) * NHASH + nh];
    size_t o = ((size_t)nh * BHD + bh) * SEQ + n;
    q_hash[o] = hq; k_hash[o] = hk;
  }
}

// bitonic argsort of 4096 keys; also inverse permutation
__global__ __launch_bounds__(1024) void k_sort(const float* q_hash,
                                               const float* k_hash, int* q_pos,
                                               int* k_pos, int* q_rev,
                                               int* k_rev) {
  __shared__ float skey[SEQ];
  __shared__ int sidx[SEQ];
  int slot = blockIdx.x;
  int isK = slot >> 6, rem = slot & 63;
  const float* h = (isK ? k_hash : q_hash) + (size_t)rem * SEQ;
  int* pos = (isK ? k_pos : q_pos) + (size_t)rem * SEQ;
  int* rev = (isK ? k_rev : q_rev) + (size_t)rem * SEQ;
  int tid = threadIdx.x;
  for (int i = tid; i < SEQ; i += 1024) { skey[i] = h[i]; sidx[i] = i; }
  __syncthreads();
  for (int kk = 2; kk <= SEQ; kk <<= 1) {
    for (int j = kk >> 1; j > 0; j >>= 1) {
      for (int i = tid; i < SEQ; i += 1024) {
        int p = i ^ j;
        if (p > i) {
          bool asc = ((i & kk) == 0);
          float a = skey[i], b = skey[p];
          bool sw = asc ? (a > b) : (a < b);
          if (sw) {
            skey[i] = b; skey[p] = a;
            int t1 = sidx[i]; sidx[i] = sidx[p]; sidx[p] = t1;
          }
        }
      }
      __syncthreads();
    }
  }
  for (int i = tid; i < SEQ; i += 1024) { int s = sidx[i]; pos[i] = s; rev[s] = i; }
}

// raw projection x @ W^T via WMMA. out: (bh, N, M) f32
__global__ __launch_bounds__(32) void k_featmap(const __bf16* __restrict__ xbf,
                                                const __bf16* __restrict__ wbf,
                                                float* __restrict__ out) {
  int bi = blockIdx.x;
  int tm = bi & 7, tn = (bi >> 3) & 255, bh = bi >> 11;
  int n0 = tn * 16, m0 = tm * 16, lane = threadIdx.x;
  v8f acc = vzero();
#pragma unroll
  for (int kk = 0; kk < EDIM; kk += 32) {
    v16bf a = frag_a_row(xbf + ((size_t)bh * SEQ + n0) * EDIM + kk, EDIM, lane);
    v16bf b = frag_bt_row(wbf + (size_t)m0 * EDIM + kk, EDIM, lane);
    acc = wmma_bf16(a, b, acc);
  }
  store_cd(acc, [&](int m, int n, float v) {
    out[((size_t)bh * SEQ + n0 + m) * MFEAT + m0 + n] = v; }, lane);
}

// q: per-row stabilize + exp; write f32 + bf16; q_ls = stab
__global__ void k_qexp(float* qp, __bf16* qpbf, const float* qn2, float* q_ls) {
  int tid = blockIdx.x * blockDim.x + threadIdx.x;
  if (tid >= BHD * SEQ) return;
  float base = -0.5f * TEMP * qn2[tid] - HALF_LOG_M;
  size_t row = (size_t)tid * MFEAT;
  float mx = -3.4e38f;
  for (int m = 0; m < MFEAT; ++m)
    mx = fmaxf(mx, qp[row + m] * SQRT_TEMP + base);
  for (int m = 0; m < MFEAT; ++m) {
    float e = __expf(qp[row + m] * SQRT_TEMP + base - mx);
    qp[row + m] = e; qpbf[row + m] = f2bf(e);
  }
  q_ls[tid] = mx;
}

// k: global (per-bh) max of log_phi
__global__ void k_kmax(const float* kp, const float* kn2, unsigned* kstab) {
  int tid = blockIdx.x * blockDim.x + threadIdx.x;
  if (tid >= BHD * SEQ) return;
  int bh = tid / SEQ;
  float base = -0.5f * TEMP * kn2[tid] - HALF_LOG_M;
  size_t row = (size_t)tid * MFEAT;
  float mx = -3.4e38f;
  for (int m = 0; m < MFEAT; ++m)
    mx = fmaxf(mx, kp[row + m] * SQRT_TEMP + base);
  atomicMax(&kstab[bh], mkey(mx));
}

// k: exp with per-bh stab; write f32 + bf16; materialize k_ls
__global__ void k_kexp(float* kp, __bf16* kpbf, const float* kn2,
                       const unsigned* kstab, float* k_ls) {
  int tid = blockIdx.x * blockDim.x + threadIdx.x;
  if (tid >= BHD * SEQ) return;
  int bh = tid / SEQ, n = tid % SEQ;
  float stab = unkey(kstab[bh]);
  if (n == 0) k_ls[bh] = stab;
  float base = -0.5f * TEMP * kn2[tid] - HALF_LOG_M;
  size_t row = (size_t)tid * MFEAT;
  for (int m = 0; m < MFEAT; ++m) {
    float e = __expf(kp[row + m] * SQRT_TEMP + base - stab);
    kp[row + m] = e; kpbf[row + m] = f2bf(e);
  }
}

// ksum[bh][m] = sum_n k_prime
__global__ void k_ksum(const float* kp, float* ksum) {
  int tid = blockIdx.x * blockDim.x + threadIdx.x;
  if (tid >= BHD * MFEAT) return;
  int bh = tid / MFEAT, m = tid % MFEAT;
  float s = 0.f;
  for (int n = 0; n < SEQ; ++n)
    s += kp[((size_t)bh * SEQ + n) * MFEAT + m];
  ksum[tid] = s;
}

// kv (stored transposed, bf16): kvbT[bh][e][m] = sum_n kp[n][m] * v[n][e]
__global__ __launch_bounds__(32) void k_kv(const __bf16* kpbf,
                                           const __bf16* vbf, __bf16* kvbT) {
  int bi = blockIdx.x;
  int tm = bi & 7, te = (bi >> 3) & 3, bh = bi >> 5;
  int m0 = tm * 16, e0 = te * 16, lane = threadIdx.x;
  v8f acc = vzero();
  for (int n0 = 0; n0 < SEQ; n0 += 32) {
    v16bf a = frag_a([&](int r, int k) {
      return kpbf[((size_t)bh * SEQ + n0 + k) * MFEAT + m0 + r]; }, lane);
    v16bf b = frag_b([&](int k, int c) {
      return vbf[((size_t)bh * SEQ + n0 + k) * EDIM + e0 + c]; }, lane);
    acc = wmma_bf16(a, b, acc);
  }
  store_cd(acc, [&](int m, int n, float v) {
    kvbT[((size_t)bh * EDIM + e0 + n) * MFEAT + m0 + m] = f2bf(v); }, lane);
}

// qk1[bh][n] = q_prime[n] . ksum[bh]
__global__ void k_qk1(const float* qp, const float* ksum, float* qk1) {
  int tid = blockIdx.x * blockDim.x + threadIdx.x;
  if (tid >= BHD * SEQ) return;
  int bh = tid / SEQ;
  size_t row = (size_t)tid * MFEAT;
  float s = 0.f;
  for (int m = 0; m < MFEAT; ++m) s += qp[row + m] * ksum[bh * MFEAT + m];
  qk1[tid] = s;
}

// qkv = q_prime @ kv  (WMMA, K=128)
__global__ __launch_bounds__(32) void k_qkv(const __bf16* qpbf,
                                            const __bf16* kvbT, float* qkv) {
  int bi = blockIdx.x;
  int tn = bi & 255, te = (bi >> 8) & 3, bh = bi >> 10;
  int n0 = tn * 16, e0 = te * 16, lane = threadIdx.x;
  v8f acc = vzero();
#pragma unroll
  for (int kk = 0; kk < MFEAT; kk += 32) {
    v16bf a = frag_a_row(qpbf + ((size_t)bh * SEQ + n0) * MFEAT + kk, MFEAT, lane);
    v16bf b = frag_bt_row(kvbT + ((size_t)bh * EDIM + e0) * MFEAT + kk, MFEAT, lane);
    acc = wmma_bf16(a, b, acc);
  }
  store_cd(acc, [&](int m, int n, float v) {
    qkv[((size_t)bh * SEQ + n0 + m) * EDIM + e0 + n] = v; }, lane);
}

// ---- bucketed exact attention: one block per (nh, bh, bucket) ----
constexpr unsigned SQ_OFF = 0;                        // 128x64 bf16
constexpr unsigned SK_OFF = SQ_OFF + BSZ * EDIM * 2;  // 16384
constexpr unsigned SV_OFF = SK_OFF + BSZ * EDIM * 2;  // 32768
constexpr unsigned SQP_OFF = SV_OFF + BSZ * EDIM * 2; // 49152, 128x128 bf16
constexpr unsigned SKP_OFF = SQP_OFF + BSZ * MFEAT * 2; // 81920
constexpr unsigned SPLS_OFF = SKP_OFF + BSZ * MFEAT * 2; // 114688
constexpr unsigned QB_OFF = SPLS_OFF + BSZ * 4;          // 115200
constexpr unsigned KB_OFF = QB_OFF + BSZ * NHASH * 4;    // 116224
constexpr unsigned QPL_OFF = KB_OFF + BSZ * NHASH * 4;   // 117248
constexpr unsigned SMEM_BYTES = QPL_OFF + BSZ * 4;       // 117760

__global__ __launch_bounds__(256) void k_bucket(
    const __bf16* qbf, const __bf16* kbf, const __bf16* vbf,
    const __bf16* qpbf, const __bf16* kpbf, const float* q_ls,
    const float* k_ls, const int* q_pos, const int* k_pos, const int* q_rev,
    const int* k_rev, float* o_ns, float* logits, float* dso) {
  __shared__ alignas(128) unsigned char smem[SMEM_BYTES];
  __bf16* sq = (__bf16*)(smem + SQ_OFF);
  __bf16* sk = (__bf16*)(smem + SK_OFF);
  __bf16* sv = (__bf16*)(smem + SV_OFF);
  __bf16* sqp = (__bf16*)(smem + SQP_OFF);
  __bf16* skp = (__bf16*)(smem + SKP_OFF);
  float* spls = (float*)(smem + SPLS_OFF);
  int* qb = (int*)(smem + QB_OFF);
  int* kb = (int*)(smem + KB_OFF);
  int* qpl = (int*)(smem + QPL_OFF);

  int bi = blockIdx.x;
  int bucket = bi & 31, bh = (bi >> 5) & 31, nh = bi >> 10;
  int t = threadIdx.x;
  size_t sbase = ((size_t)nh * BHD + bh) * SEQ + bucket * BSZ;

  if (t < BSZ) {
    int gq = q_pos[sbase + t];
    int gk = k_pos[sbase + t];
    qpl[t] = gq;
    spls[t] = q_ls[(size_t)bh * SEQ + gq] + k_ls[bh];
#pragma unroll
    for (int nh2 = 0; nh2 < NHASH; ++nh2) {
      qb[t * NHASH + nh2] = q_rev[((size_t)nh2 * BHD + bh) * SEQ + gq] >> 7;
      kb[t * NHASH + nh2] = k_rev[((size_t)nh2 * BHD + bh) * SEQ + gk] >> 7;
    }
  }

  // gather sorted rows (bf16) into LDS, 16B chunks per lane
  auto fill = [&](unsigned ldsOff, const __bf16* src, const int* pos, int epr) {
    int cpr = epr >> 3;  // 16B chunks per row
    for (int c = t; c < BSZ * cpr; c += 256) {
      int r = c / cpr, o = c % cpr;
      int g = pos[sbase + r];
      const __bf16* gp = src + ((size_t)bh * SEQ + g) * epr + o * 8;
#if USE_ASYNC
      async16(ldsOff + (unsigned)(r * epr + o * 8) * 2,
              (unsigned long long)(uintptr_t)gp);
#else
      *(v8bf*)(smem + ldsOff + (unsigned)(r * epr + o * 8) * 2) =
          *(const v8bf*)gp;
#endif
    }
  };
  fill(SQ_OFF, qbf, q_pos, EDIM);
  fill(SK_OFF, kbf, k_pos, EDIM);
  fill(SV_OFF, vbf, k_pos, EDIM);
  fill(SQP_OFF, qpbf, q_pos, MFEAT);
  fill(SKP_OFF, kpbf, k_pos, MFEAT);
#if USE_ASYNC
  wait_async();
#endif
  __syncthreads();

  int w = t >> 5, lane = t & 31;
  int h2 = (lane >> 4) & 1, nl = lane & 15;
  v8f accI[8], accP[8];
#pragma unroll
  for (int tt = 0; tt < 8; ++tt) { accI[tt] = vzero(); accP[tt] = vzero(); }

  // inner = s_q @ s_k^T  (K = 64)
#pragma unroll
  for (int kk = 0; kk < EDIM; kk += 32) {
    v16bf a = frag_a_row(sq + w * 16 * EDIM + kk, EDIM, lane);
#pragma unroll
    for (int tt = 0; tt < 8; ++tt) {
      v16bf b = frag_bt_row(sk + tt * 16 * EDIM + kk, EDIM, lane);
      accI[tt] = wmma_bf16(a, b, accI[tt]);
    }
  }
  // dots_prime = s_qp @ s_kp^T  (K = 128)
#pragma unroll
  for (int kk = 0; kk < MFEAT; kk += 32) {
    v16bf a = frag_a_row(sqp + w * 16 * MFEAT + kk, MFEAT, lane);
#pragma unroll
    for (int tt = 0; tt < 8; ++tt) {
      v16bf b = frag_bt_row(skp + tt * 16 * MFEAT + kk, MFEAT, lane);
      accP[tt] = wmma_bf16(a, b, accP[tt]);
    }
  }

  // dup correction + row max
  float rmax[8];
#pragma unroll
  for (int v = 0; v < 8; ++v) rmax[v] = -3.4e38f;
#pragma unroll
  for (int tt = 0; tt < 8; ++tt) {
#pragma unroll
    for (int v = 0; v < 8; ++v) {
      int i = w * 16 + v + 8 * h2;
      int j = tt * 16 + nl;
      float dup = (float)((qb[i * 2] == kb[j * 2]) +
                          (qb[i * 2 + 1] == kb[j * 2 + 1]));
      float in_v = accI[tt][v] * TEMP - __logf(dup);
      accI[tt][v] = in_v;
      accP[tt][v] = accP[tt][v] / dup;
      rmax[v] = fmaxf(rmax[v], in_v);
    }
  }
#pragma unroll
  for (int v = 0; v < 8; ++v) {
#pragma unroll
    for (int m = 1; m < 16; m <<= 1)
      rmax[v] = fmaxf(rmax[v], __shfl_xor(rmax[v], m, 32));
  }

  __syncthreads();              // done reading sqp before aliasing
  __bf16* sdots = sqp;          // reuse for dots matrix

  float lse[8], dsum[8];
#pragma unroll
  for (int v = 0; v < 8; ++v) {
    int i = w * 16 + v + 8 * h2;
    float pls = spls[i];
    float l = fmaxf(rmax[v], pls);
    lse[v] = l;
    float epl = __expf(pls - l);
    float s = 0.f;
#pragma unroll
    for (int tt = 0; tt < 8; ++tt) {
      float d = __expf(accI[tt][v] - l) - accP[tt][v] * epl;
      s += d;
      sdots[i * MFEAT + tt * 16 + nl] = f2bf(d);
    }
#pragma unroll
    for (int m = 1; m < 16; m <<= 1) s += __shfl_xor(s, m, 32);
    dsum[v] = s;
  }
  if (nl == 0) {
#pragma unroll
    for (int v = 0; v < 8; ++v) {
      int i = w * 16 + v + 8 * h2;
      size_t o = ((size_t)nh * BHD + bh) * SEQ + qpl[i];
      logits[o] = lse[v];
      dso[o] = dsum[v];
    }
  }
  __syncthreads();

  // so = dots @ s_v  (K = 128)
  v8f accO[4];
#pragma unroll
  for (int tt = 0; tt < 4; ++tt) accO[tt] = vzero();
#pragma unroll
  for (int kk = 0; kk < MFEAT; kk += 32) {
    v16bf a = frag_a_row(sdots + w * 16 * MFEAT + kk, MFEAT, lane);
#pragma unroll
    for (int tt = 0; tt < 4; ++tt) {
      v16bf b = frag_b([&](int k, int c) {
        return sv[(kk + k) * EDIM + tt * 16 + c]; }, lane);
      accO[tt] = wmma_bf16(a, b, accO[tt]);
    }
  }
#pragma unroll
  for (int v = 0; v < 8; ++v) {
    int i = w * 16 + v + 8 * h2;
    size_t ob = (((size_t)nh * BHD + bh) * SEQ + qpl[i]) * EDIM;
#pragma unroll
    for (int tt = 0; tt < 4; ++tt) o_ns[ob + tt * 16 + nl] = accO[tt][v];
  }
}

// combine NH branches + performer branch, normalize, pack bf16 (B,N,D)
__global__ void k_combine(const float* o_ns, const float* logits,
                          const float* dso, const float* qkv, const float* qk1,
                          const float* q_ls, const float* k_ls, __bf16* attnbf) {
  int tid = blockIdx.x * blockDim.x + threadIdx.x;
  if (tid >= BHD * SEQ) return;
  int bh = tid / SEQ, n = tid % SEQ;
  int b = bh >> 3, hh = bh & 7;
  size_t i0 = (size_t)bh * SEQ + n;
  size_t i1 = ((size_t)BHD + bh) * SEQ + n;
  float l0 = logits[i0], l1 = logits[i1];
  float ml = fmaxf(l0, l1);
  float nls = ml + __logf(__expf(l0 - ml) + __expf(l1 - ml));
  float p0 = __expf(l0 - nls), p1 = __expf(l1 - nls);
  float pls = q_ls[tid] + k_ls[bh];
  float psc = __expf(pls - nls);
  float normz = fmaxf(dso[i0] * p0 + dso[i1] * p1 + qk1[tid] * psc, 1e-6f);
  float inv = 1.f / normz;
  size_t dst = ((size_t)b * SEQ + n) * DMODEL + hh * EDIM;
#pragma unroll 4
  for (int e = 0; e < EDIM; ++e) {
    float o = o_ns[i0 * EDIM + e] * p0 + o_ns[i1 * EDIM + e] * p1 +
              qkv[(size_t)tid * EDIM + e] * psc;
    attnbf[dst + e] = f2bf(o * inv);
  }
}

// final projection: (B*N, 512) @ out_W^T (512, 64) + bias  (WMMA)
__global__ __launch_bounds__(32) void k_outproj(const __bf16* __restrict__ attnbf,
                                                const __bf16* __restrict__ wobf,
                                                const float* __restrict__ bo,
                                                float* out) {
  int bi = blockIdx.x;
  int tr = bi & 1023, tc = bi >> 10;
  int r0 = tr * 16, c0 = tc * 16, lane = threadIdx.x;
  v8f acc = vzero();
  for (int kk = 0; kk < DMODEL; kk += 32) {
    v16bf a = frag_a_row(attnbf + (size_t)r0 * DMODEL + kk, DMODEL, lane);
    v16bf b = frag_bt_row(wobf + (size_t)c0 * DMODEL + kk, DMODEL, lane);
    acc = wmma_bf16(a, b, acc);
  }
  store_cd(acc, [&](int m, int n, float v) {
    out[(size_t)(r0 + m) * EDIM + c0 + n] = v + bo[c0 + n]; }, lane);
}

// ================= launch =================
extern "C" void kernel_launch(void* const* d_in, const int* in_sizes, int n_in,
                              void* d_out, int out_size, void* d_ws,
                              size_t ws_size, hipStream_t stream) {
  const float* query = (const float*)d_in[0];
  const float* key   = (const float*)d_in[1];
  const float* value = (const float*)d_in[2];
  const float* projW = (const float*)d_in[3];
  const float* alpha = (const float*)d_in[4];
  const float* beta  = (const float*)d_in[5];
  const float* outW  = (const float*)d_in[6];
  const float* outB  = (const float*)d_in[7];
  float* out = (float*)d_out;

  size_t off = 0;
  auto alloc = [&](size_t nfloats) {
    nfloats = (nfloats + 63) & ~(size_t)63;   // keep 256B alignment
    float* p = (float*)d_ws + off;
    off += nfloats;
    return p;
  };
  auto allocb = [&](size_t nbf) { return (__bf16*)alloc((nbf + 1) / 2); };

  const size_t BN = (size_t)BHD * SEQ;
  float* q_s = alloc(BN * EDIM);
  float* k_s = alloc(BN * EDIM);
  __bf16* qbf = allocb(BN * EDIM);
  __bf16* kbf = allocb(BN * EDIM);
  __bf16* vbf = allocb(BN * EDIM);
  float* qn2 = alloc(BN);
  float* kn2 = alloc(BN);
  unsigned* maxq2 = (unsigned*)alloc(BHD);
  unsigned* maxk2 = (unsigned*)alloc(BHD);
  unsigned* kstab = (unsigned*)alloc(BHD);
  float* k_ls = alloc(BHD);
  float* q_hash = alloc(NHASH * BN);
  float* k_hash = alloc(NHASH * BN);
  int* q_pos = (int*)alloc(NHASH * BN);
  int* k_pos = (int*)alloc(NHASH * BN);
  int* q_rev = (int*)alloc(NHASH * BN);
  int* k_rev = (int*)alloc(NHASH * BN);
  float* qp = alloc(BN * MFEAT);
  float* kp = alloc(BN * MFEAT);
  __bf16* qpbf = allocb(BN * MFEAT);
  __bf16* kpbf = allocb(BN * MFEAT);
  float* q_ls = alloc(BN);
  float* ksum = alloc((size_t)BHD * MFEAT);
  float* qk1 = alloc(BN);
  __bf16* kvbT = allocb((size_t)BHD * EDIM * MFEAT);
  float* qkv = alloc(BN * EDIM);
  float* o_ns = alloc(NHASH * BN * EDIM);
  float* logits = alloc(NHASH * BN);
  float* dso = alloc(NHASH * BN);
  __bf16* attnbf = allocb((size_t)BATCH * SEQ * DMODEL);
  __bf16* wbf = allocb((size_t)MFEAT * EDIM);
  __bf16* wobf = allocb((size_t)EDIM * DMODEL);

  const int TPB = 256;
  const int NBN = (int)((BN + TPB - 1) / TPB);

  k_init<<<1, 64, 0, stream>>>(maxq2, maxk2, kstab);
  k_wconv<<<(EDIM * DMODEL + TPB - 1) / TPB, TPB, 0, stream>>>(projW, outW, wbf,
                                                               wobf);
  k_split<<<NBN, TPB, 0, stream>>>(query, key, value, q_s, k_s, qbf, kbf, vbf,
                                   qn2, kn2, maxq2, maxk2);
  k_hashes<<<NBN, TPB, 0, stream>>>(q_s, k_s, qn2, kn2, maxq2, maxk2, alpha,
                                    beta, q_hash, k_hash);
  k_sort<<<2 * NHASH * BHD, 1024, 0, stream>>>(q_hash, k_hash, q_pos, k_pos,
                                               q_rev, k_rev);
  k_featmap<<<BHD * (SEQ / 16) * (MFEAT / 16), 32, 0, stream>>>(qbf, wbf, qp);
  k_featmap<<<BHD * (SEQ / 16) * (MFEAT / 16), 32, 0, stream>>>(kbf, wbf, kp);
  k_qexp<<<NBN, TPB, 0, stream>>>(qp, qpbf, qn2, q_ls);
  k_kmax<<<NBN, TPB, 0, stream>>>(kp, kn2, kstab);
  k_kexp<<<NBN, TPB, 0, stream>>>(kp, kpbf, kn2, kstab, k_ls);
  k_ksum<<<(BHD * MFEAT) / TPB, TPB, 0, stream>>>(kp, ksum);
  k_kv<<<BHD * (MFEAT / 16) * (EDIM / 16), 32, 0, stream>>>(kpbf, vbf, kvbT);
  k_qk1<<<NBN, TPB, 0, stream>>>(qp, ksum, qk1);
  k_qkv<<<BHD * (SEQ / 16) * (EDIM / 16), 32, 0, stream>>>(qpbf, kvbT, qkv);
  k_bucket<<<NHASH * BHD * NBUK, 256, 0, stream>>>(
      qbf, kbf, vbf, qpbf, kpbf, q_ls, k_ls, q_pos, k_pos, q_rev, k_rev, o_ns,
      logits, dso);
  k_combine<<<NBN, TPB, 0, stream>>>(o_ns, logits, dso, qkv, qk1, q_ls, k_ls,
                                     attnbf);
  k_outproj<<<(BATCH * SEQ / 16) * (EDIM / 16), 32, 0, stream>>>(attnbf, wobf,
                                                                 outB, out);
}
